// QuantLinear_24721831756586
// MI455X (gfx1250) — matrix operand order repaired and checked by reading, not compiled, not run on hardware
//
#include <hip/hip_runtime.h>

// NF4-quantized linear: out[64,8192] = x[64,8192] @ dequant(W)[8192,8192]^T + bias
// MI455X / gfx1250, wave32, f16 WMMA 16x16x32 with f32 accumulation.
// Memory-bound on the packed-weight stream (134 MB): dequant path minimized to
// 1 ds_load_b32 + 1 v_pk_mul_f16 per weight-pair via a 256-entry packed-f16 LUT.

typedef __attribute__((ext_vector_type(16))) _Float16 v16h;
typedef __attribute__((ext_vector_type(8)))  _Float16 v8h;
typedef __attribute__((ext_vector_type(2)))  _Float16 v2h;
typedef __attribute__((ext_vector_type(8)))  float    v8f;
typedef __attribute__((ext_vector_type(4)))  float    v4f;
typedef __attribute__((ext_vector_type(4)))  int      v4i;

namespace {
constexpr int kIn     = 8192;   // K
constexpr int kOut    = 8192;   // N
constexpr int kM      = 64;     // rows of x
constexpr int kNTile  = 128;    // N columns per workgroup (8 waves x 16)
constexpr int kKStep  = 32;     // K per WMMA
constexpr int kKSplit = 8;      // K-dimension split across blockIdx.y
constexpr int kKChunk = kIn / kKSplit;        // 1024 K per block
constexpr int kLdsStr = 40;     // halves per LDS row: 64B data + 16B pad (16B-aligned, bank-spread)
}

__constant__ float kNf4[16] = {
  -1.0f, -0.6961928009986877f, -0.5250730514526367f, -0.39491748809814453f,
  -0.28444138169288635f, -0.18477343022823334f, -0.09105003625154495f, 0.0f,
   0.07958029955625534f, 0.16093020141124725f, 0.24611230850219726f,
   0.33791524171829224f, 0.44070982933044434f, 0.5626170039176941f,
   0.7229568362236023f, 1.0f };

__global__ __launch_bounds__(256)
void nf4_wmma_linear(const float* __restrict__ x,
                     const int*   __restrict__ packed,   // one byte (two NF4 codes) per int32
                     const float* __restrict__ absmax,   // per 64-element block
                     const float* __restrict__ bias,
                     float*       __restrict__ out)
{
    __shared__ _Float16 As[2][kM * kLdsStr];  // double-buffered 64x32 f16 chunk of x
    __shared__ unsigned pairLut[256];         // byte -> {f16 nf4[lo], f16 nf4[hi]} packed

    const int tid  = threadIdx.x;
    const int wave = tid >> 5;
    const int lane = tid & 31;
    const int nlo  = lane & 15;             // N within the wave's 16-column slab
    const int hi16 = lane >> 4;             // half-wave selector

    // 256 threads build the full byte->pair table (low nibble -> low half).
    {
        v2h pr;
        pr[0] = (_Float16)kNf4[tid & 15];
        pr[1] = (_Float16)kNf4[(tid >> 4) & 15];
        pairLut[tid] = __builtin_bit_cast(unsigned, pr);
    }

    // This lane's output column and its weight row base.
    const int  n    = blockIdx.x * kNTile + wave * 16 + nlo;
    const long nrow = (long)n * kIn;
    const float bias_v = (blockIdx.y == 0) ? bias[n] : 0.0f;

    // Cooperative A-staging addressing: 256 threads move 64x32 f32 -> f16.
    const int arow = tid >> 2;          // 0..63
    const int acol = (tid & 3) * 8;     // 0,8,16,24

    const int kBeg = blockIdx.y * kKChunk;
    const int kEnd = kBeg + kKChunk;

    v8f acc[4] = {};                    // four 16x16 M-tiles (M = 0..63)

    __syncthreads();                    // pairLut visible before first dequant

    int parity = 0;
    for (int k = kBeg; k < kEnd; k += kKStep, parity ^= 1) {
        // ---- stage x[64, k..k+31] into LDS as f16 (two b128 loads, one b128 ds store)
        const v4f* xp = reinterpret_cast<const v4f*>(x + (long)arow * kIn + k + acol);
        v4f x0 = xp[0];
        v4f x1 = xp[1];
        v8h hv;
        hv[0] = (_Float16)x0[0]; hv[1] = (_Float16)x0[1];
        hv[2] = (_Float16)x0[2]; hv[3] = (_Float16)x0[3];
        hv[4] = (_Float16)x1[0]; hv[5] = (_Float16)x1[1];
        hv[6] = (_Float16)x1[2]; hv[7] = (_Float16)x1[3];
        *reinterpret_cast<v8h*>(&As[parity][arow * kLdsStr + acol]) = hv;

        // ---- fetch + dequantize this lane's B fragment: column n, K = k0..k0+15
        // B 32x16 order used here (contiguous k, same as A): lanes 0-15 hold K 0..15, lanes 16-31 hold K 16..31.
        const int  k0   = k + hi16 * 16;
        const long flat = nrow + k0;                       // element index, multiple of 16
        const v4i* pp   = reinterpret_cast<const v4i*>(packed + (flat >> 1));
        v4i w0 = pp[0];
        v4i w1 = pp[1];
        __builtin_prefetch(reinterpret_cast<const char*>(pp) + 256, 0, 1); // 4 iters ahead
        const float scale = absmax[flat >> 6];             // whole fragment in one 64-block
        const _Float16 sh = (_Float16)scale;
        v2h s2 = { sh, sh };

        int bytes8[8] = { w0[0], w0[1], w0[2], w0[3], w1[0], w1[1], w1[2], w1[3] };
        v16h bfrag;
        #pragma unroll
        for (int i = 0; i < 8; ++i) {                      // low nibble first, then high
            v2h w = __builtin_bit_cast(v2h, pairLut[bytes8[i] & 255]);
            w = w * s2;                                    // v_pk_mul_f16
            bfrag[2 * i]     = w[0];
            bfrag[2 * i + 1] = w[1];
        }

        __syncthreads();                                   // As[parity] staged by all waves

        // ---- four WMMAs: A tiles M = t*16 .. t*16+15
        // A 16x32 layout (matches B): halves 0-15 = K kb..kb+15,
        // kb = 16*(lane>=16); lane&15 selects M row.
        const int kb = hi16 * 16;
        v8h lo8[4], hi8[4];
        #pragma unroll
        for (int t = 0; t < 4; ++t) {                      // one ds clause, one wait
            const int m = t * 16 + nlo;
            lo8[t] = *reinterpret_cast<const v8h*>(&As[parity][m * kLdsStr + kb]);
            hi8[t] = *reinterpret_cast<const v8h*>(&As[parity][m * kLdsStr + kb + 8]);
        }
        #pragma unroll
        for (int t = 0; t < 4; ++t) {                      // back-to-back WMMAs
            v16h afrag = __builtin_shufflevector(lo8[t], hi8[t],
                0, 1, 2, 3, 4, 5, 6, 7, 8, 9, 10, 11, 12, 13, 14, 15);
            acc[t] = __builtin_amdgcn_wmma_f32_16x16x32_f16(
                false, afrag, false, bfrag, (short)0, acc[t], false, false);
        }
        // No trailing barrier: next iteration writes As[parity^1]; the next
        // barrier fences re-use of this buffer (reads precede that wave's signal).
    }

    // ---- combine K-split partials. D layout: VGPR r -> M = r (lanes 0-15) / r+8 (lanes 16-31),
    // N = lane&15. bias folded in by the blockIdx.y==0 slice only.
    #pragma unroll
    for (int t = 0; t < 4; ++t) {
        #pragma unroll
        for (int r = 0; r < 8; ++r) {
            const int m = t * 16 + r + hi16 * 8;
            atomicAdd(&out[(long)m * kOut + n], acc[t][r] + bias_v);
        }
    }
}

extern "C" void kernel_launch(void* const* d_in, const int* in_sizes, int n_in,
                              void* d_out, int out_size, void* d_ws, size_t ws_size,
                              hipStream_t stream) {
    const float* x      = (const float*)d_in[0];
    const int*   packed = (const int*)  d_in[1];
    const float* absmax = (const float*)d_in[2];
    const float* bias   = (const float*)d_in[3];
    float*       out    = (float*)d_out;

    // K-split partials accumulate via atomics: zero the output first (capture-safe).
    hipMemsetAsync(out, 0, (size_t)out_size * sizeof(float), stream);

    dim3 grid(kOut / kNTile, kKSplit);   // 64 x 8 = 512 workgroups, 8 waves each
    nf4_wmma_linear<<<grid, 256, 0, stream>>>(x, packed, absmax, bias, out);
}
